// STDN_69088843924001
// MI455X (gfx1250) — compile-verified, matrix operand order checked by reading, not run on hardware
//
#include <hip/hip_runtime.h>

// ---------------------------------------------------------------------------
// Types for CDNA5 WMMA (wave32): v_wmma_f32_16x16x32_bf16
// ---------------------------------------------------------------------------
typedef __attribute__((ext_vector_type(16))) __bf16 v16bf;
typedef __attribute__((ext_vector_type(8)))  float  v8f;

union FragU { uint4 u[2]; v16bf v; };

__device__ __forceinline__ unsigned short f2bf(float x) {
  union { float f; unsigned int u; } q; q.f = x;
  unsigned int r = q.u + 0x7FFFu + ((q.u >> 16) & 1u);   // round-to-nearest-even
  return (unsigned short)(r >> 16);
}

// ---------------------------------------------------------------------------
// Weight convert + swizzle:
//   W[Cout,Cin] f32  ->  Wt[(Cin/32)][Cout][32] bf16
// so the GEMM's B-tile staging is a contiguous 32-byte copy per (n, k-half).
// ---------------------------------------------------------------------------
__global__ __launch_bounds__(256) void wconv_kernel(
    const float* __restrict__ W, unsigned short* __restrict__ Wt,
    int Cout, int Cin) {
  int idx = blockIdx.x * 256 + threadIdx.x;
  if (idx >= Cin * Cout) return;
  int k = idx / Cout;
  int o = idx - k * Cout;
  int kt = k >> 5, kr = k & 31;
  Wt[((size_t)kt * Cout + o) * 32 + kr] = f2bf(W[(size_t)o * Cin + k]);
}

// ---------------------------------------------------------------------------
// Concat: Xc[b,t,n, 0:256]=X, 256:512=TE, 512:768=SE   (float4 granularity)
// ---------------------------------------------------------------------------
__global__ __launch_bounds__(256) void concat_kernel(
    const float* __restrict__ X, const float* __restrict__ TE,
    const float* __restrict__ SE, float* __restrict__ Xc, int total4) {
  int i = blockIdx.x * 256 + threadIdx.x;
  if (i >= total4) return;
  int row = i / 192;            // 768 floats = 192 float4 per position
  int c4  = i - row * 192;
  int sel = c4 >> 6;            // 0,1,2
  int off = c4 & 63;
  const float4* src = (sel == 0) ? (const float4*)X
                    : (sel == 1) ? (const float4*)TE : (const float4*)SE;
  ((float4*)Xc)[i] = src[(size_t)row * 64 + off];
}

// ---------------------------------------------------------------------------
// bf16 WMMA GEMM:  Y[M,Cout] = act(A)[M,K] * W^T + bias
//   Wt is the pre-swizzled bf16 weight: [(K/32)][Cout][32]
//   act: identity if ssIn==nullptr, else relu(x*ssIn[c] + ssIn[K+c])
//   Tile: 128x128x32, 256 threads = 8 waves, each wave 32x64 (2x4 WMMA tiles)
//   Software-pipelined: global loads of tile k+1 overlap WMMAs of tile k.
//   Requires: M%128==0, Cout%128==0, K%32==0  (true for all layers here)
// ---------------------------------------------------------------------------
#define LSTR 40   // LDS row stride in bf16 elems (80B: 16B-aligned, bank-skewed)

__global__ __launch_bounds__(256) void wgemm_bf16(
    const float* __restrict__ A, const unsigned short* __restrict__ Wt,
    const float* __restrict__ bias, const float* __restrict__ ssIn,
    float* __restrict__ Y, int M, int K, int Cout) {
  __shared__ __align__(16) unsigned short Al[128][LSTR];
  __shared__ __align__(16) unsigned short Bl[128][LSTR];

  const int tid  = threadIdx.x;
  const int lane = tid & 31;
  const int wave = tid >> 5;
  const int wm   = wave & 3;          // 4 row-groups of 32
  const int wn   = wave >> 2;         // 2 col-groups of 64
  const int lr   = lane & 15;
  const int kb   = (lane >> 4) << 3;  // 0 / 8 : K-half select (and C-layout M ofs)

  const int n0 = blockIdx.x * 128;
  const int m0 = blockIdx.y * 128;

  v8f acc[2][4];
#pragma unroll
  for (int i = 0; i < 2; ++i)
#pragma unroll
    for (int j = 0; j < 4; ++j)
#pragma unroll
      for (int e = 0; e < 8; ++e) acc[i][j][e] = 0.0f;

  // staging assignments (one 16-elem chunk per thread for A and for B)
  const int arow  = tid >> 1;          // 0..127 (m)
  const int aoff  = (tid & 1) << 4;    // 0 / 16 (k)
  const int brow  = tid >> 1;          // 0..127 (n)
  const int bhalf = (tid & 1) << 4;    // 0 / 16 (k)

  const int nkt = K >> 5;

  // ---- register pipeline buffers ----
  float areg[16];
  uint4 breg[2];

  auto g_load = [&](int kt) {
    const float* ap = A + (size_t)(m0 + arow) * K + (kt << 5) + aoff;
#pragma unroll
    for (int j = 0; j < 16; j += 4) {
      float4 t = *(const float4*)(ap + j);
      areg[j + 0] = t.x; areg[j + 1] = t.y; areg[j + 2] = t.z; areg[j + 3] = t.w;
    }
    const uint4* bp = (const uint4*)(Wt + (((size_t)kt * Cout + n0 + brow) << 5) + bhalf);
    breg[0] = bp[0];
    breg[1] = bp[1];
    if (kt + 1 < nkt) __builtin_prefetch(ap + 32, 0, 1);
  };

  auto st_lds = [&](int kt) {
    unsigned int pk[8];
    if (ssIn) {
      const int cbase = (kt << 5) + aoff;
#pragma unroll
      for (int j = 0; j < 8; ++j) {
        float x0 = fmaxf(0.f, fmaf(areg[2 * j + 0], ssIn[cbase + 2 * j + 0],
                                   ssIn[K + cbase + 2 * j + 0]));
        float x1 = fmaxf(0.f, fmaf(areg[2 * j + 1], ssIn[cbase + 2 * j + 1],
                                   ssIn[K + cbase + 2 * j + 1]));
        pk[j] = (unsigned int)f2bf(x0) | ((unsigned int)f2bf(x1) << 16);
      }
    } else {
#pragma unroll
      for (int j = 0; j < 8; ++j)
        pk[j] = (unsigned int)f2bf(areg[2 * j]) |
                ((unsigned int)f2bf(areg[2 * j + 1]) << 16);
    }
    uint4* ad = (uint4*)&Al[arow][aoff];
    ad[0] = make_uint4(pk[0], pk[1], pk[2], pk[3]);
    ad[1] = make_uint4(pk[4], pk[5], pk[6], pk[7]);
    uint4* bd = (uint4*)&Bl[brow][bhalf];
    bd[0] = breg[0];
    bd[1] = breg[1];
  };

  g_load(0);
  for (int kt = 0; kt < nkt; ++kt) {
    st_lds(kt);
    __syncthreads();
    if (kt + 1 < nkt) g_load(kt + 1);   // overlap next-tile loads with WMMAs

    // ---- fragments + 8 WMMAs per wave ----
    v16bf af[2];
#pragma unroll
    for (int mt = 0; mt < 2; ++mt) {
      const unsigned short* p = &Al[wm * 32 + mt * 16 + lr][kb];
      FragU f;
      f.u[0] = *(const uint4*)(p);
      f.u[1] = *(const uint4*)(p + 16);
      af[mt] = f.v;
    }
#pragma unroll
    for (int nt = 0; nt < 4; ++nt) {
      const unsigned short* p = &Bl[wn * 64 + nt * 16 + lr][kb];
      FragU f;
      f.u[0] = *(const uint4*)(p);
      f.u[1] = *(const uint4*)(p + 16);
      v16bf bfr = f.v;
#pragma unroll
      for (int mt = 0; mt < 2; ++mt)
        acc[mt][nt] = __builtin_amdgcn_wmma_f32_16x16x32_bf16(
            false, af[mt], false, bfr, (short)0, acc[mt][nt], false, false);
    }
    __syncthreads();
  }

  // ---- epilogue: C/D layout -> global, + bias ----
#pragma unroll
  for (int mt = 0; mt < 2; ++mt) {
#pragma unroll
    for (int nt = 0; nt < 4; ++nt) {
      int n = n0 + wn * 64 + nt * 16 + lr;
      float bv = bias[n];
#pragma unroll
      for (int i = 0; i < 8; ++i) {
        int m = m0 + wm * 32 + mt * 16 + i + kb;   // lanes>=16 hold M=i+8
        Y[(size_t)m * Cout + n] = acc[mt][nt][i] + bv;
      }
    }
  }
}

// ---------------------------------------------------------------------------
// BN statistics (deterministic two-stage): partial per-channel sum / sumsq
// ---------------------------------------------------------------------------
__global__ __launch_bounds__(256) void stats_partial(
    const float* __restrict__ Y, float* __restrict__ part,
    int Mrows, int C, int rowsPer) {
  int c = blockIdx.x * 256 + threadIdx.x;
  if (c >= C) return;
  int r0 = blockIdx.y * rowsPer;
  int r1 = r0 + rowsPer; if (r1 > Mrows) r1 = Mrows;
  float s = 0.f, s2 = 0.f;
  for (int r = r0; r < r1; ++r) {
    float v = Y[(size_t)r * C + c];
    s += v; s2 = fmaf(v, v, s2);
  }
  part[(size_t)blockIdx.y * 2 * C + c]     = s;
  part[(size_t)blockIdx.y * 2 * C + C + c] = s2;
}

// reduce partials (fixed order) + finalize to scale/shift: relu(y*sc + sh)
__global__ __launch_bounds__(256) void bn_finalize(
    const float* __restrict__ part, int ngy, float invM, int C,
    const float* __restrict__ g, const float* __restrict__ be,
    float* __restrict__ ss) {
  int c = blockIdx.x * 256 + threadIdx.x;
  if (c >= C) return;
  float s = 0.f, s2 = 0.f;
  for (int i = 0; i < ngy; ++i) {
    s  += part[(size_t)i * 2 * C + c];
    s2 += part[(size_t)i * 2 * C + C + c];
  }
  float mu  = s * invM;
  float var = fmaxf(s2 * invM - mu * mu, 0.f);
  float rstd = rsqrtf(var + 1e-5f);
  float sc = g[c] * rstd;
  ss[c]     = sc;
  ss[C + c] = be[c] - mu * sc;
}

// ---------------------------------------------------------------------------
// Attention: one block per (b,n); 16 heads of dim 16; softmax over Sk (<=32)
// q/k/v are pre-BN GEMM outputs; BN+ReLU applied on load.
// ---------------------------------------------------------------------------
__global__ __launch_bounds__(256) void attn_kernel(
    const float* __restrict__ Yq, const float* __restrict__ Yk,
    const float* __restrict__ Yv, const float* __restrict__ ssq,
    const float* __restrict__ ssk, const float* __restrict__ ssv,
    float* __restrict__ Out, int Sq, int Sk, int N, long qBatchStride) {
  __shared__ float kl[32 * 256];
  __shared__ float vl[32 * 256];
  const int n = blockIdx.x % N;
  const int b = blockIdx.x / N;
  const int tid = threadIdx.x;

  for (int i = tid; i < Sk * 256; i += 256) {
    const int sk = i >> 8, c = i & 255;
    const size_t idx = (((size_t)b * Sk + sk) * N + n) * 256 + c;
    kl[i] = fmaxf(0.f, fmaf(Yk[idx], ssk[c], ssk[256 + c]));
    vl[i] = fmaxf(0.f, fmaf(Yv[idx], ssv[c], ssv[256 + c]));
  }
  __syncthreads();

  const float scale = 0.25f;  // 1/sqrt(NUM_D=16)
  for (int it = tid; it < Sq * 16; it += 256) {
    const int h = it & 15, qi = it >> 4;
    const size_t qbase = (size_t)b * qBatchStride +
                         (((size_t)qi * N + n) * 256) + h * 16;
    float qv[16];
#pragma unroll
    for (int j = 0; j < 16; ++j) {
      const int c = h * 16 + j;
      qv[j] = fmaxf(0.f, fmaf(Yq[qbase + j], ssq[c], ssq[256 + c]));
    }
    float lg[32];
    float mx = -3.0e38f;
    for (int sk = 0; sk < Sk; ++sk) {
      const float* kp = &kl[sk * 256 + h * 16];
      float d = 0.f;
#pragma unroll
      for (int j = 0; j < 16; ++j) d = fmaf(qv[j], kp[j], d);
      d *= scale; lg[sk] = d; mx = fmaxf(mx, d);
    }
    float sum = 0.f;
    for (int sk = 0; sk < Sk; ++sk) { float e = __expf(lg[sk] - mx); lg[sk] = e; sum += e; }
    const float inv = 1.0f / sum;
    float ov[16];
#pragma unroll
    for (int j = 0; j < 16; ++j) ov[j] = 0.f;
    for (int sk = 0; sk < Sk; ++sk) {
      const float p = lg[sk] * inv;
      const float* vp = &vl[sk * 256 + h * 16];
#pragma unroll
      for (int j = 0; j < 16; ++j) ov[j] = fmaf(p, vp[j], ov[j]);
    }
    const size_t obase = (((size_t)(b * Sq + qi) * N + n) * 256) + h * 16;
#pragma unroll
    for (int j = 0; j < 16; ++j) Out[obase + j] = ov[j];
  }
}

// ---------------------------------------------------------------------------
// Final: out = X + relu(Y*sc + sh)
// ---------------------------------------------------------------------------
__global__ __launch_bounds__(256) void final_kernel(
    const float* __restrict__ X, const float* __restrict__ Y,
    const float* __restrict__ ss, float* __restrict__ out, int total) {
  int i = blockIdx.x * 256 + threadIdx.x;
  if (i >= total) return;
  int c = i & 255;
  out[i] = X[i] + fmaxf(0.f, fmaf(Y[i], ss[c], ss[256 + c]));
}

// ---------------------------------------------------------------------------
// Host orchestration
// ---------------------------------------------------------------------------
extern "C" void kernel_launch(void* const* d_in, const int* in_sizes, int n_in,
                              void* d_out, int out_size, void* d_ws, size_t ws_size,
                              hipStream_t stream) {
  (void)in_sizes; (void)n_in; (void)out_size;
  const int B = 16, T = 12, N = 500, S = 32;
  const int Mq0 = S * N;          // 16000  (Ir is batch-broadcast of I)
  const int Mkv0 = B * T * N;     // 96000
  const int Mo0 = B * S * N;      // 256000
  const int Mq1 = Mkv0, Mkv1 = Mo0, Mo1 = Mkv0;

  const float* X  = (const float*)d_in[0];
  const float* TE = (const float*)d_in[1];
  const float* SE = (const float*)d_in[2];
  const float* I  = (const float*)d_in[3];
  // layers: 0 m0q, 1 m0k, 2 m0v, 3 m0o, 4 m1q, 5 m1k, 6 m1v, 7 m1o
  const float *W[8], *Bi[8], *G[8], *Be[8];
  for (int l = 0; l < 8; ++l) {
    W[l]  = (const float*)d_in[4 + l * 4 + 0];
    Bi[l] = (const float*)d_in[4 + l * 4 + 1];
    G[l]  = (const float*)d_in[4 + l * 4 + 2];
    Be[l] = (const float*)d_in[4 + l * 4 + 3];
  }
  const int Cin[8]  = {768, 768, 768, 256, 768, 768, 768, 256};
  const int Cout[8] = {256, 256, 256, 768, 256, 256, 256, 256};

  // ---- workspace layout ----
  char* wsb = (char*)d_ws;
  size_t off = 0;
  auto alloc = [&](size_t bytes) -> void* {
    off = (off + 255) & ~(size_t)255;
    void* p = wsb + off;
    off += bytes;
    return p;
  };
  float* Xc   = (float*)alloc((size_t)Mkv0 * 768 * 4);
  float* yq0  = (float*)alloc((size_t)Mq0  * 256 * 4);
  float* yk0  = (float*)alloc((size_t)Mkv0 * 256 * 4);
  float* yv0  = (float*)alloc((size_t)Mkv0 * 256 * 4);
  float* ao0  = (float*)alloc((size_t)Mo0  * 256 * 4);
  float* yH   = (float*)alloc((size_t)Mo0  * 768 * 4);
  float* yv1  = (float*)alloc((size_t)Mkv1 * 256 * 4);
  float* yo1  = (float*)alloc((size_t)Mo1  * 256 * 4);
  float* yq1  = yk0;   // free after attn0
  float* yk1  = ao0;   // free after m0o GEMM
  float* ao1  = yv0;   // free after attn0
  float* part = (float*)alloc((size_t)256 * 2 * 768 * 4);
  unsigned short* Wt[8];
  for (int l = 0; l < 8; ++l)
    Wt[l] = (unsigned short*)alloc((size_t)Cin[l] * Cout[l] * 2);
  float* ssb[8];
  for (int l = 0; l < 8; ++l) ssb[l] = (float*)alloc((size_t)2 * Cout[l] * 4);
  if (off > ws_size) return;  // insufficient workspace: bail deterministically

  // ---- helpers ----
  auto run_gemm = [&](const float* A, const unsigned short* Wtp, const float* b,
                      const float* ssIn, float* Y, int M, int K, int C) {
    dim3 g(C / 128, M / 128);
    wgemm_bf16<<<g, 256, 0, stream>>>(A, Wtp, b, ssIn, Y, M, K, C);
  };
  auto run_bn = [&](const float* Y, int M, int C, int layer) {
    const int rowsPer = 1000;
    int gy = (M + rowsPer - 1) / rowsPer;
    dim3 gr((C + 255) / 256, gy);
    stats_partial<<<gr, 256, 0, stream>>>(Y, part, M, C, rowsPer);
    bn_finalize<<<(C + 255) / 256, 256, 0, stream>>>(
        part, gy, 1.0f / (float)M, C, G[layer], Be[layer], ssb[layer]);
  };

  // ---- 1) weights -> bf16, transposed + K-tile swizzled ----
  for (int l = 0; l < 8; ++l) {
    int tot = Cin[l] * Cout[l];
    wconv_kernel<<<(tot + 255) / 256, 256, 0, stream>>>(W[l], Wt[l], Cout[l], Cin[l]);
  }

  // ---- 2) Xc = concat(X, TE, SE) ----
  {
    int total4 = Mkv0 * 192;
    concat_kernel<<<(total4 + 255) / 256, 256, 0, stream>>>(X, TE, SE, Xc, total4);
  }

  // ---- 3) MAB0 ----
  run_gemm(I,  Wt[0], Bi[0], nullptr, yq0, Mq0,  768, 256); run_bn(yq0, Mq0,  256, 0);
  run_gemm(Xc, Wt[1], Bi[1], nullptr, yk0, Mkv0, 768, 256); run_bn(yk0, Mkv0, 256, 1);
  run_gemm(Xc, Wt[2], Bi[2], nullptr, yv0, Mkv0, 768, 256); run_bn(yv0, Mkv0, 256, 2);
  attn_kernel<<<B * N, 256, 0, stream>>>(yq0, yk0, yv0, ssb[0], ssb[1], ssb[2],
                                         ao0, /*Sq=*/32, /*Sk=*/12, N,
                                         /*qBatchStride=*/0L);
  run_gemm(ao0, Wt[3], Bi[3], nullptr, yH, Mo0, 256, 768); run_bn(yH, Mo0, 768, 3);

  // ---- 4) MAB1 ----
  run_gemm(Xc, Wt[4], Bi[4], nullptr, yq1, Mq1,  768, 256); run_bn(yq1, Mq1,  256, 4);
  run_gemm(yH, Wt[5], Bi[5], ssb[3], yk1, Mkv1, 768, 256); run_bn(yk1, Mkv1, 256, 5);
  run_gemm(yH, Wt[6], Bi[6], ssb[3], yv1, Mkv1, 768, 256); run_bn(yv1, Mkv1, 256, 6);
  attn_kernel<<<B * N, 256, 0, stream>>>(yq1, yk1, yv1, ssb[4], ssb[5], ssb[6],
                                         ao1, /*Sq=*/12, /*Sk=*/32, N,
                                         /*qBatchStride=*/(long)12 * N * 256);
  run_gemm(ao1, Wt[7], Bi[7], nullptr, yo1, Mo1, 256, 256); run_bn(yo1, Mo1, 256, 7);

  // ---- 5) out = X + relu(BN(yo1)) ----
  {
    int total = Mo1 * 256;  // 24,576,000 == B*T*N*256
    final_kernel<<<(total + 255) / 256, 256, 0, stream>>>(X, yo1, ssb[7],
                                                          (float*)d_out, total);
  }
}
